// EdgeDegreeEmbedding_49546742727014
// MI455X (gfx1250) — matrix operand order repaired and checked by reading, not compiled, not run on hardware
//
#include <hip/hip_runtime.h>
#include <hip/hip_bf16.h>

typedef __bf16 v16bf __attribute__((ext_vector_type(16)));
typedef float  v8f   __attribute__((ext_vector_type(8)));
typedef float  v2f   __attribute__((ext_vector_type(2)));

#define SPHERE_C 128
#define NUM_COEFF 25
#define M_ALL 19
#define M0 5
#define WAVES 4

// ---------------------------------------------------------------------------
// Weight swizzle: f32 row-major [K=128][N] -> bf16 in WMMA B-fragment order:
//   [n_tile][kk][lane][16 bf16], element j of lane's v16bf = W[kk*32 + (lane>>4)*8 + (j&7) + (j>>3)*16][t*16 + (lane&15)]
// so a B fragment is one contiguous 32-byte load per lane.
// ---------------------------------------------------------------------------
__global__ void swizzle_w_kernel(const float* __restrict__ W, __bf16* __restrict__ dst,
                                 int N, int total) {
    int idx = blockIdx.x * blockDim.x + threadIdx.x;
    if (idx >= total) return;
    int j    = idx & 15;
    int lane = (idx >> 4) & 31;
    int kk   = (idx >> 9) & 3;
    int t    = idx >> 11;
    int k = kk * 32 + (lane >> 4) * 8 + (j & 7) + (j >> 3) * 16;
    int n = t * 16 + (lane & 15);
    dst[idx] = (__bf16)W[(size_t)k * N + n];
}

// A fragment (bf16 16x32) from per-wave LDS tile [16 rows][128 cols]:
// lane l (row = l&15): halves j=0..7 -> K = base8+j, j=8..15 -> K = base8+16+(j-8),
// base8 = (l>=16)?8:0  (per ISA 16-bit A 16x32 layout).
__device__ __forceinline__ v16bf load_afragA(const __bf16* Ast, int lane, int kk) {
    int row = lane & 15;
    int lh  = lane >> 4;
    const __bf16* p = Ast + row * 128 + kk * 32 + lh * 8;
    v16bf a;
#pragma unroll
    for (int j = 0; j < 8; ++j) a[j] = p[j];
#pragma unroll
    for (int j = 0; j < 8; ++j) a[8 + j] = p[16 + j];
    return a;
}

// One MLP layer: A(16x128, bf16 in LDS) @ W(128x128, swizzled bf16) ->
// +bias -> LayerNorm -> SiLU -> written back into the same LDS tile (bf16).
__device__ __forceinline__ void mlp_layer128(const __bf16* __restrict__ Wsw,
                                             const float* __restrict__ bias,
                                             const float* __restrict__ g,
                                             const float* __restrict__ be,
                                             __bf16* Ast, int lane) {
    int lh = lane >> 4, ln = lane & 15;
    v16bf a[4];
#pragma unroll
    for (int kk = 0; kk < 4; ++kk) a[kk] = load_afragA(Ast, lane, kk);

    v8f acc[8];
#pragma unroll
    for (int t = 0; t < 8; ++t) {
        v8f c = {};
#pragma unroll
        for (int kk = 0; kk < 4; ++kk) {
            v16bf b = *(const v16bf*)(Wsw + (size_t)(((t * 4 + kk) * 32 + lane) * 16));
            c = __builtin_amdgcn_wmma_f32_16x16x32_bf16(false, a[kk], false, b,
                                                        (short)0, c, false, false);
        }
        acc[t] = c;
    }
    // bias
#pragma unroll
    for (int t = 0; t < 8; ++t) {
        float bv = bias[t * 16 + ln];
#pragma unroll
        for (int r = 0; r < 8; ++r) acc[t][r] += bv;
    }
    // LayerNorm + SiLU, per C-tile row (row = r + lh*8), reduce across 16-lane half
#pragma unroll
    for (int r = 0; r < 8; ++r) {
        float s1 = 0.f, s2 = 0.f;
#pragma unroll
        for (int t = 0; t < 8; ++t) { float v = acc[t][r]; s1 += v; s2 += v * v; }
#pragma unroll
        for (int m = 1; m < 16; m <<= 1) {
            s1 += __shfl_xor(s1, m, 32);
            s2 += __shfl_xor(s2, m, 32);
        }
        float mu  = s1 * (1.0f / 128.0f);
        float var = s2 * (1.0f / 128.0f) - mu * mu;
        float rs  = rsqrtf(var + 1e-6f);
#pragma unroll
        for (int t = 0; t < 8; ++t) {
            int col = t * 16 + ln;
            float h = (acc[t][r] - mu) * rs * g[col] + be[col];
            h = h / (1.0f + __expf(-h));   // SiLU
            acc[t][r] = h;
        }
    }
    // write back (all A-fragment reads already completed; DS in-order per wave)
    asm volatile("s_wait_dscnt 0" ::: "memory");
#pragma unroll
    for (int r = 0; r < 8; ++r) {
        int row = r + lh * 8;
#pragma unroll
        for (int t = 0; t < 8; ++t) Ast[row * 128 + t * 16 + ln] = (__bf16)acc[t][r];
    }
    asm volatile("s_wait_dscnt 0" ::: "memory");
}

// ---------------------------------------------------------------------------
// Main fused kernel: one wave = 16 edges. MLP (bf16 WMMA) -> per-16-col chunk:
// GEMM3 tiles (k=0..4) -> wigner einsum (fp32 WMMA 16x16x4) -> scaled atomics.
// ---------------------------------------------------------------------------
__global__ void __launch_bounds__(WAVES * 32)
edge_degree_kernel(const float* __restrict__ x_edge,
                   const int* __restrict__ edge_index,
                   const float* __restrict__ wigner,
                   const float* __restrict__ envelope,
                   const int* __restrict__ node_offset_p,
                   const float* __restrict__ b1, const float* __restrict__ g1, const float* __restrict__ be1,
                   const float* __restrict__ b2, const float* __restrict__ g2, const float* __restrict__ be2,
                   const float* __restrict__ b3,
                   const __bf16* __restrict__ W1sw,
                   const __bf16* __restrict__ W2sw,
                   const __bf16* __restrict__ W3sw,
                   float* __restrict__ out, int E) {
    __shared__ __bf16 s_ast[WAVES][16 * 128];        // per-wave 16x128 activation tile
    __shared__ float  s_h3[WAVES][M0][16][17];       // per-wave GEMM3 staging (padded)

    const int lane = threadIdx.x & 31;
    const int wave = threadIdx.x >> 5;
    const int lh = lane >> 4, ln = lane & 15;
    const int e0 = (blockIdx.x * WAVES + wave) * 16;
    const int node_offset = node_offset_p[0];
    __bf16* Ast = &s_ast[wave][0];

    // Stage x_edge tile -> LDS (f32 -> bf16), coalesced, zero-padded past E
#pragma unroll
    for (int i = 0; i < 64; ++i) {
        int idx = i * 32 + lane;
        int row = idx >> 7, col = idx & 127;
        float v = (e0 + row < E) ? x_edge[(size_t)(e0 + row) * 128 + col] : 0.0f;
        Ast[idx] = (__bf16)v;
    }
    asm volatile("s_wait_dscnt 0" ::: "memory");

    mlp_layer128(W1sw, b1, g1, be1, Ast, lane);
    mlp_layer128(W2sw, b2, g2, be2, Ast, lane);

    // Keep h2 A-fragments resident for all GEMM3 chunks
    v16bf a2[4];
#pragma unroll
    for (int kk = 0; kk < 4; ++kk) a2[kk] = load_afragA(Ast, lane, kk);

    for (int ct = 0; ct < 8; ++ct) {              // 16-wide channel chunk of SPHERE_C
        // GEMM3: h3[k][16 edges][16 cols] for k = 0..4 (W3 col n = (k*8+ct)*16 ..)
#pragma unroll
        for (int k = 0; k < M0; ++k) {
            v8f c = {};
#pragma unroll
            for (int kk = 0; kk < 4; ++kk) {
                v16bf b = *(const v16bf*)(W3sw + (size_t)((((k * 8 + ct) * 4 + kk) * 32 + lane) * 16));
                c = __builtin_amdgcn_wmma_f32_16x16x32_bf16(false, a2[kk], false, b,
                                                            (short)0, c, false, false);
            }
            float bv = b3[(k * 8 + ct) * 16 + ln];
#pragma unroll
            for (int r = 0; r < 8; ++r)
                s_h3[wave][k][r + lh * 8][ln] = c[r] + bv;
        }
        asm volatile("s_wait_dscnt 0" ::: "memory");

        // Per-edge rotation: emb[j, c] = sum_{k<5} wig[j,k] * h3[k,c]  (fp32 WMMA)
        for (int e = 0; e < 16; ++e) {
            int ge = e0 + e;
            if (ge >= E) break;                    // wave-uniform
            float scale = envelope[ge] * (1.0f / 16.0f);  // env / RESCALE
            int tgt = edge_index[E + ge] - node_offset;

            // B fragments (4x16 f32): V0 = K=lh*2, V1 = K=lh*2+1, N = ln
            v2f bk0, bk1;
            bk0[0] = s_h3[wave][lh * 2 + 0][e][ln];
            bk0[1] = s_h3[wave][lh * 2 + 1][e][ln];
            bk1[0] = (lh == 0) ? s_h3[wave][4][e][ln] : 0.0f;  // K=4 valid, K>=5 zero
            bk1[1] = 0.0f;

#pragma unroll
            for (int m = 0; m < 2; ++m) {          // j-tiles: rows 0..15, 16..24
                int jr = m * 16 + ln;
                bool jv = jr < NUM_COEFF;
                const float* wp = wigner + ((size_t)ge * NUM_COEFF + (jv ? jr : 0)) * M_ALL;
                v2f a0, a1;
                a0[0] = jv ? wp[lh * 2 + 0] : 0.0f;
                a0[1] = jv ? wp[lh * 2 + 1] : 0.0f;
                a1[0] = (jv && lh == 0) ? wp[4] : 0.0f;
                a1[1] = 0.0f;

                v8f d = {};
                d = __builtin_amdgcn_wmma_f32_16x16x4_f32(false, a0, false, bk0,
                                                          (short)0, d, false, false);
                d = __builtin_amdgcn_wmma_f32_16x16x4_f32(false, a1, false, bk1,
                                                          (short)0, d, false, false);
#pragma unroll
                for (int r = 0; r < 8; ++r) {
                    int j = m * 16 + r + lh * 8;
                    if (j < NUM_COEFF)
                        atomicAdd(out + ((size_t)tgt * NUM_COEFF + j) * SPHERE_C + ct * 16 + ln,
                                  d[r] * scale);
                }
            }
        }
    }
}

extern "C" void kernel_launch(void* const* d_in, const int* in_sizes, int n_in,
                              void* d_out, int out_size, void* d_ws, size_t ws_size,
                              hipStream_t stream) {
    const float* x        = (const float*)d_in[0];
    const float* x_edge   = (const float*)d_in[1];
    const int*   eidx     = (const int*)d_in[2];
    const float* wigner   = (const float*)d_in[3];
    const float* envelope = (const float*)d_in[4];
    const int*   noff     = (const int*)d_in[5];
    const float* W1 = (const float*)d_in[6];  const float* b1 = (const float*)d_in[7];
    const float* g1 = (const float*)d_in[8];  const float* be1 = (const float*)d_in[9];
    const float* W2 = (const float*)d_in[10]; const float* b2 = (const float*)d_in[11];
    const float* g2 = (const float*)d_in[12]; const float* be2 = (const float*)d_in[13];
    const float* W3 = (const float*)d_in[14]; const float* b3 = (const float*)d_in[15];
    float* out = (float*)d_out;

    const int E = in_sizes[1] / 128;

    // Workspace: swizzled bf16 weights
    __bf16* W1sw = (__bf16*)d_ws;                  // 128*128 elems
    __bf16* W2sw = W1sw + 128 * 128;               // 128*128 elems
    __bf16* W3sw = W2sw + 128 * 128;               // 128*640 elems

    const int t1 = 128 * 128, t3 = 128 * 640;
    swizzle_w_kernel<<<(t1 + 255) / 256, 256, 0, stream>>>(W1, W1sw, 128, t1);
    swizzle_w_kernel<<<(t1 + 255) / 256, 256, 0, stream>>>(W2, W2sw, 128, t1);
    swizzle_w_kernel<<<(t3 + 255) / 256, 256, 0, stream>>>(W3, W3sw, 640, t3);

    // out = x, then edge kernel scatter-adds env*emb/16 on top
    hipMemcpyAsync(out, x, (size_t)in_sizes[0] * sizeof(float),
                   hipMemcpyDeviceToDevice, stream);

    const int edges_per_block = WAVES * 16;
    const int blocks = (E + edges_per_block - 1) / edges_per_block;
    edge_degree_kernel<<<blocks, WAVES * 32, 0, stream>>>(
        x_edge, eidx, wigner, envelope, noff,
        b1, g1, be1, b2, g2, be2, b3,
        W1sw, W2sw, W3sw, out, E);
}
